// OptimizedSNN_32590211842113
// MI455X (gfx1250) — matrix operand order repaired and checked by reading, not compiled, not run on hardware
//
#include <hip/hip_runtime.h>
#include <hip/hip_bf16.h>
#include <stdint.h>

typedef __attribute__((ext_vector_type(16))) _Float16 v16h;
typedef __attribute__((ext_vector_type(8)))  _Float16 v8h;
typedef __attribute__((ext_vector_type(8)))  float    v8f;
typedef __attribute__((ext_vector_type(4)))  float    vf4;

#define T_STEPS 100
#define BATCH   1024
#define NIN     784
#define NH      400
#define NO      10
#define KB1     25    // ceil(784/32)
#define NT1     25    // 400/16
#define KB2     13    // ceil(400/32)
#define EPSBN   1e-5f
#define KEEP_SCALE (1.0f/0.7f)

// ---------------- threefry2x32 with key (0,42) == jax.random.key(42) -------
__device__ __forceinline__ unsigned rotl32(unsigned x, int d) {
  return (x << d) | (x >> (32 - d));
}

__device__ __forceinline__ void threefry2x32_42(unsigned c0, unsigned c1,
                                                unsigned& o0, unsigned& o1) {
  const unsigned ks0 = 0u, ks1 = 42u, ks2 = 0u ^ 42u ^ 0x1BD11BDAu;
  unsigned x0 = c0 + ks0, x1 = c1 + ks1;
#define TFR(r) { x0 += x1; x1 = rotl32(x1, r); x1 ^= x0; }
  TFR(13) TFR(15) TFR(26) TFR(6)   x0 += ks1; x1 += ks2 + 1u;
  TFR(17) TFR(29) TFR(16) TFR(24)  x0 += ks2; x1 += ks0 + 2u;
  TFR(13) TFR(15) TFR(26) TFR(6)   x0 += ks0; x1 += ks1 + 3u;
  TFR(17) TFR(29) TFR(16) TFR(24)  x0 += ks1; x1 += ks2 + 4u;
  TFR(13) TFR(15) TFR(26) TFR(6)   x0 += ks2; x1 += ks0 + 5u;
#undef TFR
  o0 = x0; o1 = x1;
}

// ---------------- one-shot weight conversion into WMMA B-operand layout ----
// B operand (32x16, K x N, f16): lane l holds N = ntile*16 + (l&15), the 16
// halves are K = kb*32 + 8*(l>=16) + {0..7, 16..23}  (ISA 7.12.2 / 7.12.5 B layout).
__global__ void prep_weights(const float* __restrict__ W1,
                             const float* __restrict__ W2,
                             _Float16* __restrict__ w1sw,
                             _Float16* __restrict__ w2sw) {
  int tid = blockIdx.x * blockDim.x + threadIdx.x;
  const int E1 = NT1 * KB1 * 512;
  if (tid < E1) {
    int j  = tid & 15;
    int l  = (tid >> 4) & 31;
    int kb = (tid >> 9) % KB1;
    int nt = tid / (512 * KB1);
    int k = kb * 32 + ((l & 16) ? 8 : 0) + ((j < 8) ? j : j + 8);
    int n = nt * 16 + (l & 15);
    float v = (k < NIN) ? W1[n * NIN + k] : 0.f;
    w1sw[tid] = (_Float16)v;
  } else {
    int t2 = tid - E1;
    if (t2 < KB2 * 512) {
      int j  = t2 & 15;
      int l  = (t2 >> 4) & 31;
      int kb = t2 >> 9;
      int k = kb * 32 + ((l & 16) ? 8 : 0) + ((j < 8) ? j : j + 8);
      float v = ((l & 15) < NO && k < NH) ? W2[(l & 15) * NH + k] : 0.f;
      w2sw[t2] = (_Float16)v;
    }
  }
}

// ---------------- zero all carried state (deterministic per call) ----------
__global__ void init_state(float* __restrict__ v1, float* __restrict__ v2,
                           float* __restrict__ bn, float* __restrict__ outsum) {
  int tid = blockIdx.x * blockDim.x + threadIdx.x;
  if (tid < BATCH * NH)  v1[tid] = 0.f;
  if (tid < BATCH * 16)  v2[tid] = 0.f;
  if (tid < 2 * 2 * NH)  bn[tid] = 0.f;
  if (tid < BATCH * NO)  outsum[tid] = 0.f;
}

// ---------------- layer-1 GEMM: H = X_t @ W1^T via v_wmma_f32_16x16x32_f16 -
// Each wave: one 16-row M tile x 5 N tiles (16x80), K swept in 32-chunks.
// A layout per ISA: lane holds row M=lane&15; 16 halves are two contiguous
// 8-K runs at offsets {0|8} and {16|24} depending on lane<16.
__global__ void __launch_bounds__(256)
gemm1_wmma(const float* __restrict__ X, const _Float16* __restrict__ w1sw,
           float* __restrict__ H, float* __restrict__ bnsum) {
  const int lane  = threadIdx.x & 31;
  const int wave  = threadIdx.x >> 5;
  const int mtile = blockIdx.x * 8 + wave;   // 0..63
  const int ng    = blockIdx.y;              // 0..4 (5 N-tiles each)
  const int row   = mtile * 16 + (lane & 15);
  const float* __restrict__ xr = X + (size_t)row * NIN;
  const v16h* __restrict__ w1v = (const v16h*)w1sw;
  const int koff = (lane & 16) ? 8 : 0;

  v8f acc[5] = {};

  for (int kb = 0; kb < KB1; kb++) {
    const int k0 = kb * 32 + koff;
    __builtin_prefetch(xr + k0 + 64, 0, 0);
    v16h a;
    if (k0 + 8 <= NIN) {                 // K chunk [k0, k0+8)
      vf4 f0 = *(const vf4*)(xr + k0);
      vf4 f1 = *(const vf4*)(xr + k0 + 4);
#pragma unroll
      for (int r = 0; r < 4; r++) a[r] = (_Float16)f0[r];
#pragma unroll
      for (int r = 0; r < 4; r++) a[4 + r] = (_Float16)f1[r];
    } else {
#pragma unroll
      for (int r = 0; r < 8; r++) a[r] = (_Float16)0.f;
    }
    if (k0 + 24 <= NIN) {                // K chunk [k0+16, k0+24)
      vf4 f0 = *(const vf4*)(xr + k0 + 16);
      vf4 f1 = *(const vf4*)(xr + k0 + 20);
#pragma unroll
      for (int r = 0; r < 4; r++) a[8 + r] = (_Float16)f0[r];
#pragma unroll
      for (int r = 0; r < 4; r++) a[12 + r] = (_Float16)f1[r];
    } else {
#pragma unroll
      for (int r = 0; r < 8; r++) a[8 + r] = (_Float16)0.f;
    }
#pragma unroll
    for (int i = 0; i < 5; i++) {
      const int nt = ng * 5 + i;
      v16h b = w1v[(nt * KB1 + kb) * 32 + lane];
      acc[i] = __builtin_amdgcn_wmma_f32_16x16x32_f16(
          false, a, false, b, (short)0, acc[i], false, false);
    }
  }

  // C layout: VGPR r -> row mtile*16 + r + 8*(lane>=16), col nbase + (lane&15)
  const int r0 = mtile * 16 + ((lane & 16) ? 8 : 0);
#pragma unroll
  for (int i = 0; i < 5; i++) {
    const int n = (ng * 5 + i) * 16 + (lane & 15);
    float s = 0.f, sq = 0.f;
#pragma unroll
    for (int r = 0; r < 8; r++) {
      float v = acc[i][r];
      H[(size_t)(r0 + r) * NH + n] = v;
      s  += v;
      sq += v * v;
    }
    atomicAdd(&bnsum[n], s);
    atomicAdd(&bnsum[NH + n], sq);
  }
}

// ---------------- BN + LIF1 + threefry dropout ----------------------------
__global__ void __launch_bounds__(256)
lif1_bn_drop(const float* __restrict__ H, const float* __restrict__ bnsum,
             float* __restrict__ bnnext, const float* __restrict__ gamma1,
             const float* __restrict__ beta1, float* __restrict__ v1,
             _Float16* __restrict__ s1h, float* __restrict__ hidden_out,
             int t) {
  int tid = blockIdx.x * blockDim.x + threadIdx.x;  // < 409600
  int n = tid % NH;
  float mean = bnsum[n] * (1.f / BATCH);
  float var  = bnsum[NH + n] * (1.f / BATCH) - mean * mean;
  float inv  = rsqrtf(var + EPSBN);
  float hbn  = gamma1[n] * (H[tid] - mean) * inv + beta1[n];
  float v = v1[tid];
  v = 0.5f * (v + hbn);                 // decay_input, tau=2
  float s = (v >= 1.0f) ? 1.f : 0.f;    // threshold 1.0
  v1[tid] = v * (1.f - s);              // hard reset

  // jax.random.uniform(key(42), (T,B,400)) >= 0.3, inverted-dropout scale.
  unsigned i = (unsigned)t * (unsigned)(BATCH * NH) + (unsigned)tid;
  const unsigned half = (unsigned)T_STEPS * BATCH * NH / 2u;  // 20480000
  unsigned p = (i < half) ? i : (i - half);
  unsigned r0, r1;
  threefry2x32_42(p, p + half, r0, r1);
  unsigned bits = (i < half) ? r0 : r1;
  float u = __uint_as_float(0x3f800000u | (bits >> 9)) - 1.0f;
  float mask = (u >= 0.3f) ? KEEP_SCALE : 0.f;

  float sd = s * mask;
  hidden_out[tid] = sd;                 // hidden_spikes output (= s1d)
  s1h[tid] = (_Float16)sd;              // f16 feed for layer-2 GEMM

  if (tid < 2 * NH) bnnext[tid] = 0.f;  // clear other BN parity for t+1
}

// ---------------- layer-2 GEMM + BN + LIF2 (single workgroup) -------------
__global__ void __launch_bounds__(256)
gemm2_lif2(const _Float16* __restrict__ s1h, const _Float16* __restrict__ w2sw,
           const float* __restrict__ gamma2, const float* __restrict__ beta2,
           float* __restrict__ v2, float* __restrict__ outsum) {
  __shared__ float ssum[16];
  __shared__ float ssq[16];
  const int lane = threadIdx.x & 31;
  const int wave = threadIdx.x >> 5;
  const v16h* __restrict__ w2v = (const v16h*)w2sw;
  const int koff = (lane & 16) ? 8 : 0;

  v8f acc[8] = {};

  for (int kb = 0; kb < KB2; kb++) {
    v16h bm = w2v[kb * 32 + lane];
    const int k0 = kb * 32 + koff;
#pragma unroll
    for (int i = 0; i < 8; i++) {
      const int mtile = wave + i * 8;            // 64 M-tiles over 8 waves
      const int row = mtile * 16 + (lane & 15);
      const _Float16* __restrict__ ar = s1h + (size_t)row * NH;
      v16h a;
      if (k0 + 8 <= NH) {
        v8h c = *(const v8h*)(ar + k0);
#pragma unroll
        for (int r = 0; r < 8; r++) a[r] = c[r];
      } else {
#pragma unroll
        for (int r = 0; r < 8; r++) a[r] = (_Float16)0.f;
      }
      if (k0 + 24 <= NH) {
        v8h c = *(const v8h*)(ar + k0 + 16);
#pragma unroll
        for (int r = 0; r < 8; r++) a[8 + r] = c[r];
      } else {
#pragma unroll
        for (int r = 0; r < 8; r++) a[8 + r] = (_Float16)0.f;
      }
      acc[i] = __builtin_amdgcn_wmma_f32_16x16x32_f16(
          false, a, false, bm, (short)0, acc[i], false, false);
    }
  }

  // Batch statistics over all 1024 rows via LDS reduction.
  if (threadIdx.x < 16) { ssum[threadIdx.x] = 0.f; ssq[threadIdx.x] = 0.f; }
  __syncthreads();
  float ps = 0.f, pq = 0.f;
#pragma unroll
  for (int i = 0; i < 8; i++)
#pragma unroll
    for (int r = 0; r < 8; r++) {
      float v = acc[i][r];
      ps += v;
      pq += v * v;
    }
  const int n = lane & 15;
  atomicAdd(&ssum[n], ps);
  atomicAdd(&ssq[n], pq);
  __syncthreads();

  float mean = ssum[n] * (1.f / BATCH);
  float var  = ssq[n] * (1.f / BATCH) - mean * mean;
  float inv  = rsqrtf(var + EPSBN);
  float g  = (n < NO) ? gamma2[n] : 0.f;
  float bt = (n < NO) ? beta2[n]  : 0.f;

  const int rbase = (lane & 16) ? 8 : 0;
#pragma unroll
  for (int i = 0; i < 8; i++) {
    const int mtile = wave + i * 8;
#pragma unroll
    for (int r = 0; r < 8; r++) {
      int brow = mtile * 16 + rbase + r;
      float o = g * (acc[i][r] - mean) * inv + bt;
      float v = v2[brow * 16 + n];
      v = 0.5f * (v + o);
      float s = (v >= 1.0f) ? 1.f : 0.f;
      v2[brow * 16 + n] = v * (1.f - s);
      if (n < NO) outsum[brow * NO + n] += s;   // out_spikes.sum(axis=0)
    }
  }
}

// ---------------- launcher -------------------------------------------------
extern "C" void kernel_launch(void* const* d_in, const int* in_sizes, int n_in,
                              void* d_out, int out_size, void* d_ws,
                              size_t ws_size, hipStream_t stream) {
  (void)in_sizes; (void)n_in; (void)out_size; (void)ws_size;
  const float* x  = (const float*)d_in[0];
  const float* W1 = (const float*)d_in[1];
  const float* g1 = (const float*)d_in[2];
  const float* b1 = (const float*)d_in[3];
  const float* W2 = (const float*)d_in[4];
  const float* g2 = (const float*)d_in[5];
  const float* b2 = (const float*)d_in[6];

  char* ws = (char*)d_ws;
  const size_t o_v1  = 0;                          // 1024*400*4
  const size_t o_v2  = o_v1  + 1638400;            // 1024*16*4
  const size_t o_H   = o_v2  + 65536;              // 1024*400*4
  const size_t o_s1h = o_H   + 1638400;            // 1024*400*2
  const size_t o_w1  = o_s1h + 819200;             // 25*25*512*2
  const size_t o_w2  = o_w1  + 640000;             // 13*512*2
  const size_t o_bn  = o_w2  + 13312;              // 2 parities * 800 * 4

  float*    v1    = (float*)(ws + o_v1);
  float*    v2    = (float*)(ws + o_v2);
  float*    H     = (float*)(ws + o_H);
  _Float16* s1h   = (_Float16*)(ws + o_s1h);
  _Float16* w1sw  = (_Float16*)(ws + o_w1);
  _Float16* w2sw  = (_Float16*)(ws + o_w2);
  float*    bn    = (float*)(ws + o_bn);

  float* outsum = (float*)d_out;                     // [1024,10]
  float* hidden = (float*)d_out + BATCH * NO;        // [100,1024,400]

  const int prep_elems = NT1 * KB1 * 512 + KB2 * 512;
  prep_weights<<<(prep_elems + 255) / 256, 256, 0, stream>>>(W1, W2, w1sw, w2sw);
  init_state<<<(BATCH * NH + 255) / 256, 256, 0, stream>>>(v1, v2, bn, outsum);

  for (int t = 0; t < T_STEPS; t++) {
    float* bncur = bn + (size_t)(t & 1) * (2 * NH);
    float* bnnxt = bn + (size_t)((t + 1) & 1) * (2 * NH);
    gemm1_wmma<<<dim3(8, 5), 256, 0, stream>>>(
        x + (size_t)t * BATCH * NIN, w1sw, H, bncur);
    lif1_bn_drop<<<(BATCH * NH) / 256, 256, 0, stream>>>(
        H, bncur, bnnxt, g1, b1, v1, s1h, hidden + (size_t)t * BATCH * NH, t);
    gemm2_lif2<<<1, 256, 0, stream>>>(s1h, w2sw, g2, b2, v2, outsum);
  }
}
